// EgnnLayer_56135222559302
// MI455X (gfx1250) — compile-verified, matrix operand order checked by reading, not run on hardware
//
#include <hip/hip_runtime.h>

#define DHID 128
#define WAVES 4

typedef __attribute__((ext_vector_type(16))) _Float16 v16h;
typedef __attribute__((ext_vector_type(8)))  _Float16 v8h;
typedef __attribute__((ext_vector_type(8)))  float    v8f;

__device__ __forceinline__ float silu_f(float t) {
    return t * (1.0f / (1.0f + __expf(-t)));
}

__device__ __forceinline__ v8f vzero8() {
    v8f z;
#pragma unroll
    for (int j = 0; j < 8; ++j) z[j] = 0.0f;
    return z;
}

// B fragment: pre-swizzled in workspace. Lane l holds column n = nt*16+(l&15),
// 8 contiguous K halfs at kt*32+(l>>4)*8 and 8 at +16 (ISA 16-bit B 32x16 layout).
__device__ __forceinline__ v16h load_b_frag(const _Float16* __restrict__ w,
                                            int kt, int nt, int lane) {
    return *(const v16h*)(w + (((size_t)(kt * 8 + nt)) * 32 + lane) * 16);
}

// A fragment from an LDS-resident 16x128 f16 row-major tile.
__device__ __forceinline__ v16h lds_a_frag(const _Float16* sbuf, int kt, int lane) {
    const int m  = lane & 15;
    const int ks = kt * 32 + ((lane >> 4) << 3);
    v8h lo = *(const v8h*)(sbuf + m * DHID + ks);
    v8h hi = *(const v8h*)(sbuf + m * DHID + ks + 16);
    v16h r;
#pragma unroll
    for (int j = 0; j < 8; ++j) { r[j] = lo[j]; r[8 + j] = hi[j]; }
    return r;
}

// 8 contiguous f16 feature values for one edge row starting at feature index k.
// Feature vector: [h_src(0..127) | h_dst(128..255) | 6 scalars (256..261) | pad to 288]
__device__ __forceinline__ v8h feat8(const float* __restrict__ h,
                                     int srow, int drow, const float* esc, int k) {
    v8h o;
    if (k < 128) {
        const float4* p = (const float4*)(h + (size_t)srow * DHID + k);
        float4 a = p[0], b = p[1];
        o[0] = (_Float16)a.x; o[1] = (_Float16)a.y; o[2] = (_Float16)a.z; o[3] = (_Float16)a.w;
        o[4] = (_Float16)b.x; o[5] = (_Float16)b.y; o[6] = (_Float16)b.z; o[7] = (_Float16)b.w;
    } else if (k < 256) {
        const float4* p = (const float4*)(h + (size_t)drow * DHID + (k - 128));
        float4 a = p[0], b = p[1];
        o[0] = (_Float16)a.x; o[1] = (_Float16)a.y; o[2] = (_Float16)a.z; o[3] = (_Float16)a.w;
        o[4] = (_Float16)b.x; o[5] = (_Float16)b.y; o[6] = (_Float16)b.z; o[7] = (_Float16)b.w;
    } else {
#pragma unroll
        for (int j = 0; j < 8; ++j) {
            int kk = k + j;
            float v = (kk < 262) ? esc[kk - 256] : 0.0f;
            o[j] = (_Float16)v;
        }
    }
    return o;
}

// 8 contiguous f16 values of the node-MLP input [h | m_i] for one node row.
__device__ __forceinline__ v8h node8(const float* __restrict__ h,
                                     const float* __restrict__ macc,
                                     int node, float invd, int k) {
    v8h o;
    if (k < 128) {
        const float4* p = (const float4*)(h + (size_t)node * DHID + k);
        float4 a = p[0], b = p[1];
        o[0] = (_Float16)a.x; o[1] = (_Float16)a.y; o[2] = (_Float16)a.z; o[3] = (_Float16)a.w;
        o[4] = (_Float16)b.x; o[5] = (_Float16)b.y; o[6] = (_Float16)b.z; o[7] = (_Float16)b.w;
    } else {
        const float4* p = (const float4*)(macc + (size_t)node * DHID + (k - 128));
        float4 a = p[0], b = p[1];
        o[0] = (_Float16)(a.x * invd); o[1] = (_Float16)(a.y * invd);
        o[2] = (_Float16)(a.z * invd); o[3] = (_Float16)(a.w * invd);
        o[4] = (_Float16)(b.x * invd); o[5] = (_Float16)(b.y * invd);
        o[6] = (_Float16)(b.z * invd); o[7] = (_Float16)(b.w * invd);
    }
    return o;
}

#define WMMA_F16(A, B, C) \
    __builtin_amdgcn_wmma_f32_16x16x32_f16(false, (A), false, (B), (short)0, (C), false, false)

// ---------------------------------------------------------------------------
// Prep: zero segment-sum accumulators
__global__ void zero_kernel(float* __restrict__ p, unsigned long long n) {
    unsigned long long i = (unsigned long long)blockIdx.x * blockDim.x + threadIdx.x;
    if (i < n) p[i] = 0.0f;
}

// Prep: convert W[Kreal x 128] f32 (row-major, [in,out]) into swizzled f16 B-fragments,
// zero-padded to KT*32 rows. One thread per (kt, nt, lane) entry = 16 halfs.
__global__ void prep_wfrag_kernel(const float* __restrict__ W, int Kreal, int KT,
                                  _Float16* __restrict__ dstp) {
    int idx = blockIdx.x * blockDim.x + threadIdx.x;
    if (idx >= KT * 8 * 32) return;
    int lane = idx & 31;
    int tile = idx >> 5;
    int nt = tile & 7, kt = tile >> 3;
    int n  = nt * 16 + (lane & 15);
    int ks = kt * 32 + ((lane >> 4) << 3);
    v16h v;
#pragma unroll
    for (int j = 0; j < 8; ++j) {
        int k = ks + j;
        v[j] = (k < Kreal) ? (_Float16)W[(size_t)k * DHID + n] : (_Float16)0.0f;
    }
#pragma unroll
    for (int j = 0; j < 8; ++j) {
        int k = ks + 16 + j;
        v[8 + j] = (k < Kreal) ? (_Float16)W[(size_t)k * DHID + n] : (_Float16)0.0f;
    }
    *(v16h*)(dstp + (size_t)idx * 16) = v;
}

// ---------------------------------------------------------------------------
// Edge kernel: one wave handles one 16-edge tile (E = 400000 is a multiple of 16).
__global__ __launch_bounds__(WAVES * 32) void egnn_edge_kernel(
    const float* __restrict__ h, const float* __restrict__ x,
    const int* __restrict__ src, const int* __restrict__ dst,
    const float* __restrict__ dist, const float* __restrict__ bpp,
    const float* __restrict__ msa, const float* __restrict__ chem,
    const float* __restrict__ rel, const float* __restrict__ cbm,
    const _Float16* __restrict__ fW1, const float* __restrict__ b1,
    const _Float16* __restrict__ fW2, const float* __restrict__ b2,
    const _Float16* __restrict__ fC1, const float* __restrict__ cb1,
    const float* __restrict__ cW2, const float* __restrict__ cb2,
    float* __restrict__ m_acc, float* __restrict__ delta,
    float* __restrict__ cnt, int E)
{
    __shared__ _Float16 s_buf[WAVES][16 * DHID];
    __shared__ float    s_esc[WAVES][16 * 6];
    __shared__ int      s_src[WAVES][16];
    __shared__ int      s_dst[WAVES][16];
    __shared__ float    s_coef[WAVES][16];

    const int wid  = threadIdx.x >> 5;
    const int lane = threadIdx.x & 31;
    const int tile = blockIdx.x * WAVES + wid;
    const int e0   = tile * 16;
    if (e0 >= E) return;

    if (lane < 16) {
        const int e = e0 + lane;
        s_src[wid][lane] = src[e];
        s_dst[wid][lane] = dst[e];
        const float dd = dist[e];
        s_esc[wid][lane * 6 + 0] = dd * dd;
        s_esc[wid][lane * 6 + 1] = bpp[e];
        s_esc[wid][lane * 6 + 2] = msa[e];
        s_esc[wid][lane * 6 + 3] = chem[e];
        s_esc[wid][lane * 6 + 4] = rel[e];
        s_esc[wid][lane * 6 + 5] = cbm[e];
        s_coef[wid][lane] = 0.0f;
    }
    __builtin_amdgcn_wave_barrier();

    const int m    = lane & 15;
    const int hi16 = lane >> 4;
    const int srow = s_src[wid][m];
    const int drow = s_dst[wid][m];
    const float* esc = &s_esc[wid][m * 6];

    // ---- GEMM 1: feat[16 x 288(pad)] @ msg_W1 -> l1[16 x 128]
    v8f acc[8];
#pragma unroll
    for (int nt = 0; nt < 8; ++nt) acc[nt] = vzero8();
#pragma unroll
    for (int kt = 0; kt < 9; ++kt) {
        const int k0 = kt * 32 + hi16 * 8;
        v8h lo = feat8(h, srow, drow, esc, k0);
        v8h hv = feat8(h, srow, drow, esc, k0 + 16);
        v16h a;
#pragma unroll
        for (int j = 0; j < 8; ++j) { a[j] = lo[j]; a[8 + j] = hv[j]; }
#pragma unroll
        for (int nt = 0; nt < 8; ++nt) {
            v16h b = load_b_frag(fW1, kt, nt, lane);
            acc[nt] = WMMA_F16(a, b, acc[nt]);
        }
    }
    // SiLU -> LDS f16 (row-major 16x128)
#pragma unroll
    for (int nt = 0; nt < 8; ++nt) {
        const int n = nt * 16 + m;
        const float bias = b1[n];
#pragma unroll
        for (int v = 0; v < 8; ++v) {
            const int row = v + hi16 * 8;
            s_buf[wid][row * DHID + n] = (_Float16)silu_f(acc[nt][v] + bias);
        }
    }
    __builtin_amdgcn_wave_barrier();

    // ---- GEMM 2: l1 @ msg_W2 -> m_ij[16 x 128]
    v8f acc2[8];
#pragma unroll
    for (int nt = 0; nt < 8; ++nt) acc2[nt] = vzero8();
#pragma unroll
    for (int kt = 0; kt < 4; ++kt) {
        v16h a = lds_a_frag(&s_buf[wid][0], kt, lane);
#pragma unroll
        for (int nt = 0; nt < 8; ++nt) {
            v16h b = load_b_frag(fW2, kt, nt, lane);
            acc2[nt] = WMMA_F16(a, b, acc2[nt]);
        }
    }
    __builtin_amdgcn_wave_barrier();
    // SiLU; scatter-add into m_acc (segment_sum); keep f16 copy in LDS for coef GEMM
#pragma unroll
    for (int nt = 0; nt < 8; ++nt) {
        const int n = nt * 16 + m;
        const float bias = b2[n];
#pragma unroll
        for (int v = 0; v < 8; ++v) {
            const int row = v + hi16 * 8;
            const float t = silu_f(acc2[nt][v] + bias);
            s_buf[wid][row * DHID + n] = (_Float16)t;
            atomicAdd(&m_acc[(size_t)s_src[wid][row] * DHID + n], t);
        }
    }
    __builtin_amdgcn_wave_barrier();

    // ---- GEMM 3: m_ij @ c_W1 -> l3[16 x 128]
    v8f acc3[8];
#pragma unroll
    for (int nt = 0; nt < 8; ++nt) acc3[nt] = vzero8();
#pragma unroll
    for (int kt = 0; kt < 4; ++kt) {
        v16h a = lds_a_frag(&s_buf[wid][0], kt, lane);
#pragma unroll
        for (int nt = 0; nt < 8; ++nt) {
            v16h b = load_b_frag(fC1, kt, nt, lane);
            acc3[nt] = WMMA_F16(a, b, acc3[nt]);
        }
    }
    // coef[m] = tanh(sum_k silu(l3[m,k]+cb1[k]) * cW2[k] + cb2)
    float part[8];
#pragma unroll
    for (int v = 0; v < 8; ++v) part[v] = 0.0f;
#pragma unroll
    for (int nt = 0; nt < 8; ++nt) {
        const int n = nt * 16 + m;
        const float bias = cb1[n];
        const float w = cW2[n];
#pragma unroll
        for (int v = 0; v < 8; ++v) part[v] += silu_f(acc3[nt][v] + bias) * w;
    }
#pragma unroll
    for (int v = 0; v < 8; ++v) atomicAdd(&s_coef[wid][v + hi16 * 8], part[v]);
    __builtin_amdgcn_wave_barrier();

    if (lane < 16) {
        const float c = tanhf(s_coef[wid][lane] + cb2[0]);
        const int s = s_src[wid][lane];
        const int d = s_dst[wid][lane];
#pragma unroll
        for (int j = 0; j < 3; ++j) {
            const float df = x[(size_t)s * 3 + j] - x[(size_t)d * 3 + j];
            atomicAdd(&delta[(size_t)s * 3 + j], df * c);
        }
        atomicAdd(&cnt[s], 1.0f);
    }
}

// ---------------------------------------------------------------------------
// Node kernel: one wave per 16-node tile (N = 50000 is a multiple of 16).
__global__ __launch_bounds__(WAVES * 32) void egnn_node_kernel(
    const float* __restrict__ h, const float* __restrict__ x,
    const _Float16* __restrict__ fW1, const float* __restrict__ b1,
    const _Float16* __restrict__ fW2, const float* __restrict__ b2,
    const float* __restrict__ m_acc, const float* __restrict__ delta,
    const float* __restrict__ cnt,
    float* __restrict__ out_h, float* __restrict__ out_x, int N)
{
    __shared__ _Float16 s_buf[WAVES][16 * DHID];
    __shared__ float    s_den[WAVES][16];

    const int wid  = threadIdx.x >> 5;
    const int lane = threadIdx.x & 31;
    const int tile = blockIdx.x * WAVES + wid;
    const int nb   = tile * 16;
    if (nb >= N) return;

    if (lane < 16) s_den[wid][lane] = fmaxf(cnt[nb + lane], 1.0f);
    __builtin_amdgcn_wave_barrier();

    const int m    = lane & 15;
    const int hi16 = lane >> 4;
    const int node = nb + m;
    const float invd = 1.0f / s_den[wid][m];

    // ---- GEMM 1: [h | m_i][16 x 256] @ h_W1 -> l1[16 x 128]
    v8f acc[8];
#pragma unroll
    for (int nt = 0; nt < 8; ++nt) acc[nt] = vzero8();
#pragma unroll
    for (int kt = 0; kt < 8; ++kt) {
        const int k0 = kt * 32 + hi16 * 8;
        v8h lo = node8(h, m_acc, node, invd, k0);
        v8h hv = node8(h, m_acc, node, invd, k0 + 16);
        v16h a;
#pragma unroll
        for (int j = 0; j < 8; ++j) { a[j] = lo[j]; a[8 + j] = hv[j]; }
#pragma unroll
        for (int nt = 0; nt < 8; ++nt) {
            v16h b = load_b_frag(fW1, kt, nt, lane);
            acc[nt] = WMMA_F16(a, b, acc[nt]);
        }
    }
#pragma unroll
    for (int nt = 0; nt < 8; ++nt) {
        const int n = nt * 16 + m;
        const float bias = b1[n];
#pragma unroll
        for (int v = 0; v < 8; ++v) {
            const int row = v + hi16 * 8;
            s_buf[wid][row * DHID + n] = (_Float16)silu_f(acc[nt][v] + bias);
        }
    }
    __builtin_amdgcn_wave_barrier();

    // ---- GEMM 2: l1 @ h_W2 ; residual add -> h_out
    v8f acc2[8];
#pragma unroll
    for (int nt = 0; nt < 8; ++nt) acc2[nt] = vzero8();
#pragma unroll
    for (int kt = 0; kt < 4; ++kt) {
        v16h a = lds_a_frag(&s_buf[wid][0], kt, lane);
#pragma unroll
        for (int nt = 0; nt < 8; ++nt) {
            v16h b = load_b_frag(fW2, kt, nt, lane);
            acc2[nt] = WMMA_F16(a, b, acc2[nt]);
        }
    }
#pragma unroll
    for (int nt = 0; nt < 8; ++nt) {
        const int n = nt * 16 + m;
        const float bias = b2[n];
#pragma unroll
        for (int v = 0; v < 8; ++v) {
            const int row = v + hi16 * 8;
            const int nd  = nb + row;
            out_h[(size_t)nd * DHID + n] = h[(size_t)nd * DHID + n] + acc2[nt][v] + bias;
        }
    }

    if (lane < 16) {
        const int nd = nb + lane;
        const float id = 1.0f / s_den[wid][lane];
#pragma unroll
        for (int j = 0; j < 3; ++j)
            out_x[(size_t)nd * 3 + j] = x[(size_t)nd * 3 + j] + delta[(size_t)nd * 3 + j] * id;
    }
}

// ---------------------------------------------------------------------------
extern "C" void kernel_launch(void* const* d_in, const int* in_sizes, int n_in,
                              void* d_out, int out_size, void* d_ws, size_t ws_size,
                              hipStream_t stream) {
    (void)n_in; (void)out_size; (void)ws_size;
    const float* h         = (const float*)d_in[0];
    const float* x         = (const float*)d_in[1];
    const int*   src       = (const int*)d_in[2];
    const int*   dstn      = (const int*)d_in[3];
    const float* distances = (const float*)d_in[4];
    const float* bpp       = (const float*)d_in[5];
    const float* msa       = (const float*)d_in[6];
    const float* chem      = (const float*)d_in[7];
    const float* rel       = (const float*)d_in[8];
    const float* cbm       = (const float*)d_in[9];
    const float* msg_W1    = (const float*)d_in[10];
    const float* msg_b1    = (const float*)d_in[11];
    const float* msg_W2    = (const float*)d_in[12];
    const float* msg_b2    = (const float*)d_in[13];
    const float* h_W1      = (const float*)d_in[14];
    const float* h_b1      = (const float*)d_in[15];
    const float* h_W2      = (const float*)d_in[16];
    const float* h_b2      = (const float*)d_in[17];
    const float* c_W1      = (const float*)d_in[18];
    const float* c_b1      = (const float*)d_in[19];
    const float* c_W2      = (const float*)d_in[20];
    const float* c_b2      = (const float*)d_in[21];

    const int N = in_sizes[0] / DHID;
    const int E = in_sizes[2];

    // Workspace layout: [m_acc N*128 | delta N*3 | cnt N] f32, then f16 weight frags.
    char*  ws     = (char*)d_ws;
    float* m_acc  = (float*)ws;
    float* delta  = m_acc + (size_t)N * DHID;
    float* cnt    = delta + (size_t)N * 3;
    const size_t accFloats = (size_t)N * DHID + (size_t)N * 3 + (size_t)N;
    const size_t fragOff   = ((accFloats * 4 + 63) / 64) * 64;
    _Float16* f_msgW1 = (_Float16*)(ws + fragOff);       // 9*8*512 halfs
    _Float16* f_msgW2 = f_msgW1 + 9 * 8 * 512;           // 4*8*512
    _Float16* f_hW1   = f_msgW2 + 4 * 8 * 512;           // 8*8*512
    _Float16* f_hW2   = f_hW1   + 8 * 8 * 512;           // 4*8*512
    _Float16* f_cW1   = f_hW2   + 4 * 8 * 512;           // 4*8*512

    // 1) zero segment-sum accumulators
    {
        unsigned long long n = (unsigned long long)accFloats;
        int blocks = (int)((n + 255) / 256);
        zero_kernel<<<blocks, 256, 0, stream>>>(m_acc, n);
    }
    // 2) pre-swizzle weights into f16 B-fragment layout (KT*256 threads each)
    prep_wfrag_kernel<<<9, 256, 0, stream>>>(msg_W1, 262, 9, f_msgW1);
    prep_wfrag_kernel<<<4, 256, 0, stream>>>(msg_W2, 128, 4, f_msgW2);
    prep_wfrag_kernel<<<8, 256, 0, stream>>>(h_W1,   256, 8, f_hW1);
    prep_wfrag_kernel<<<4, 256, 0, stream>>>(h_W2,   128, 4, f_hW2);
    prep_wfrag_kernel<<<4, 256, 0, stream>>>(c_W1,   128, 4, f_cW1);

    // 3) fused edge pipeline (msg MLP + coef MLP + scatter-adds)
    const int etiles = (E + 15) / 16;
    egnn_edge_kernel<<<(etiles + WAVES - 1) / WAVES, WAVES * 32, 0, stream>>>(
        h, x, src, dstn, distances, bpp, msa, chem, rel, cbm,
        f_msgW1, msg_b1, f_msgW2, msg_b2, f_cW1, c_b1, c_W2, c_b2,
        m_acc, delta, cnt, E);

    // 4) fused node update
    float* out_h = (float*)d_out;
    float* out_x = out_h + (size_t)N * DHID;
    const int ntiles = (N + 15) / 16;
    egnn_node_kernel<<<(ntiles + WAVES - 1) / WAVES, WAVES * 32, 0, stream>>>(
        h, x, f_hW1, h_b1, f_hW2, h_b2, m_acc, delta, cnt, out_h, out_x, N);
}
